// LlamaAttentionLSQInteger_84026740179620
// MI455X (gfx1250) — matrix operand order repaired and checked by reading, not compiled, not run on hardware
//
#include <hip/hip_runtime.h>
#include <math.h>

// ---------------------------------------------------------------------------
// CDNA5 (gfx1250) wave32 WMMA implementation of Llama attention + LSQ quant.
// v3: GEMM 128x64 block tile, 2x2 accumulators/wave (4 WMMA per k-step);
// async global->LDS tile staging (global_load_async_to_lds_b128 + ASYNCcnt)
// with LDS double buffering; fp16 operands staged once; b128 everywhere.
// ---------------------------------------------------------------------------

typedef __attribute__((ext_vector_type(16))) _Float16 v16h;
typedef __attribute__((ext_vector_type(8)))  float    v8f;

#define BATCH 4
#define S_LEN 1024
#define HDIM  4096
#define NHQ   32
#define NKVH  8
#define HEADD 128

// Forward LSQ: s_sc == s in forward, xq == round(clip(x/s,qn,qp)).
__device__ __forceinline__ float lsq_fwd(float x, float s, int sym) {
  float qn = sym ? -128.0f : 0.0f;
  float qp = sym ?  127.0f : 255.0f;
  float xs = fminf(fmaxf(x / s, qn), qp);
  return rintf(xs) * s;   // RNE matches jnp.round
}

// A fragment (16x32 f16, MxK). LDS layout: lds[m*stride + k], stride in halves.
__device__ __forceinline__ v16h frag_a_16x32(const _Float16* lds, int stride, int lane) {
  union { v16h v; unsigned int u[8]; } f;
  int m    = lane & 15;
  int koff = (lane < 16) ? 0 : 16;
  const _Float16* base = lds + m * stride;
#pragma unroll
  for (int i = 0; i < 4; ++i) {
    f.u[i]     = *reinterpret_cast<const unsigned int*>(base + koff + 2 * i);
    f.u[4 + i] = *reinterpret_cast<const unsigned int*>(base + koff + 8 + 2 * i);
  }
  return f.v;
}

// B fragment (32x16 f16, KxN). LDS stores B transposed: lds[n*stride + k].
__device__ __forceinline__ v16h frag_b_32x16(const _Float16* lds, int stride, int lane) {
  union { v16h v; unsigned int u[8]; } f;
  int n    = lane & 15;
  int koff = (lane < 16) ? 0 : 16;
  const _Float16* base = lds + n * stride + koff;
#pragma unroll
  for (int i = 0; i < 8; ++i)
    f.u[i] = *reinterpret_cast<const unsigned int*>(base + 2 * i);
  return f.v;
}

// Async copy of 16 bytes global -> LDS (per-lane addresses), ASYNCcnt-tracked.
__device__ __forceinline__ void async_g2l_b128(void* lds_ptr, const void* gptr) {
  unsigned lds = (unsigned)(size_t)lds_ptr;              // low 32b = LDS addr
  unsigned long long ga = (unsigned long long)(size_t)gptr;
  asm volatile("global_load_async_to_lds_b128 %0, %1, off"
               :: "v"(lds), "v"(ga) : "memory");
}
__device__ __forceinline__ void async_wait0() {
  asm volatile("s_wait_asynccnt 0x0" ::: "memory");
}

// ---------------------------------------------------------------------------
// Bulk fp32 -> fp16 conversion, b128 reads / b64 writes.
// ---------------------------------------------------------------------------
__global__ __launch_bounds__(256) void f32_to_f16_kernel(
    const float* __restrict__ in, _Float16* __restrict__ out, int n4)
{
  int i = blockIdx.x * blockDim.x + threadIdx.x;
  if (i >= n4) return;
  float4 v = reinterpret_cast<const float4*>(in)[i];
  union { _Float16 h[4]; float2 f2; } u;
  u.h[0] = (_Float16)v.x; u.h[1] = (_Float16)v.y;
  u.h[2] = (_Float16)v.z; u.h[3] = (_Float16)v.w;
  reinterpret_cast<float2*>(out)[i] = u.f2;
}

// ---------------------------------------------------------------------------
// C[M,N] = A[M,K] * W[N,K]^T  (A, W fp16), LSQ epilogue.
// Block tile 128x64: 8 waves in 4x2, each wave a 32x32 output patch
// (2x2 accumulators, 4 WMMAs per 32-deep k-step). LDS double buffered,
// tiles staged with async global->LDS b128 copies; one barrier per k-step.
// ---------------------------------------------------------------------------
__global__ __launch_bounds__(256) void gemm_h_wmma(
    const _Float16* __restrict__ A, const _Float16* __restrict__ W,
    float* __restrict__ C, _Float16* __restrict__ Ch,
    int M, int N, int K, const float* __restrict__ s_ptr, int sym)
{
  __shared__ _Float16 As[2][128][40];   // 40-half stride: 80B rows (16B mult)
  __shared__ _Float16 Ws[2][64][40];
  int tid  = threadIdx.x;
  int lane = tid & 31;
  int w    = tid >> 5;
  int waveRow = w >> 1;        // 0..3 -> 32-tall M slice
  int waveCol = w & 1;         // 0..1 -> 32-wide N slice
  int blockM = blockIdx.x * 128;
  int blockN = blockIdx.y * 64;

  int lr = tid >> 2;           // 0..63: tile row loaded by this thread
  int lc = (tid & 3) * 8;      // 8-half chunk within 32-deep k slice

  auto load_tile = [&](int buf, int k0) {
    async_g2l_b128(&As[buf][lr][lc],
                   A + (size_t)(blockM + lr) * K + k0 + lc);
    async_g2l_b128(&As[buf][lr + 64][lc],
                   A + (size_t)(blockM + lr + 64) * K + k0 + lc);
    async_g2l_b128(&Ws[buf][lr][lc],
                   W + (size_t)(blockN + lr) * K + k0 + lc);
  };

  v8f acc00 = {}, acc01 = {}, acc10 = {}, acc11 = {};
  const int nk = K >> 5;
  load_tile(0, 0);
  for (int kt = 0; kt < nk; ++kt) {
    async_wait0();             // own async copies for buf (kt&1) are done
    __syncthreads();           // everyone's copies are done
    if (kt + 1 < nk) {
      load_tile((kt + 1) & 1, (kt + 1) << 5);
      if (kt + 2 < nk) {       // prefetch k-tile after next into L2
        __builtin_prefetch(A + (size_t)(blockM + lr) * K + ((kt + 2) << 5) + lc, 0, 1);
        __builtin_prefetch(W + (size_t)(blockN + lr) * K + ((kt + 2) << 5) + lc, 0, 1);
      }
    }
    int b = kt & 1;
    v16h a0 = frag_a_16x32(&As[b][waveRow * 32][0],      40, lane);
    v16h a1 = frag_a_16x32(&As[b][waveRow * 32 + 16][0], 40, lane);
    v16h b0 = frag_b_32x16(&Ws[b][waveCol * 32][0],      40, lane);
    v16h b1 = frag_b_32x16(&Ws[b][waveCol * 32 + 16][0], 40, lane);
    acc00 = __builtin_amdgcn_wmma_f32_16x16x32_f16(false, a0, false, b0,
                                                   (short)0, acc00, false, false);
    acc01 = __builtin_amdgcn_wmma_f32_16x16x32_f16(false, a0, false, b1,
                                                   (short)0, acc01, false, false);
    acc10 = __builtin_amdgcn_wmma_f32_16x16x32_f16(false, a1, false, b0,
                                                   (short)0, acc10, false, false);
    acc11 = __builtin_amdgcn_wmma_f32_16x16x32_f16(false, a1, false, b1,
                                                   (short)0, acc11, false, false);
  }

  float s = *s_ptr;
  int n0 = blockN + waveCol * 32 + (lane & 15);
  int m0 = blockM + waveRow * 32 + ((lane < 16) ? 0 : 8);
#pragma unroll
  for (int r = 0; r < 8; ++r) {
    float v00 = lsq_fwd(acc00[r], s, sym);
    float v01 = lsq_fwd(acc01[r], s, sym);
    float v10 = lsq_fwd(acc10[r], s, sym);
    float v11 = lsq_fwd(acc11[r], s, sym);
    size_t o0 = (size_t)(m0 + r) * N + n0;
    size_t o1 = (size_t)(m0 + 16 + r) * N + n0;
    if (C) {
      C[o0] = v00;  C[o0 + 16] = v01;
      C[o1] = v10;  C[o1 + 16] = v11;
    }
    if (Ch) {
      Ch[o0] = (_Float16)v00;  Ch[o0 + 16] = (_Float16)v01;
      Ch[o1] = (_Float16)v10;  Ch[o1 + 16] = (_Float16)v11;
    }
  }
}

// ---------------------------------------------------------------------------
// Rotate-half RoPE: reads fp32 projection, writes fp16 operand buffer.
// ---------------------------------------------------------------------------
__global__ void rope_h_kernel(const float* __restrict__ x, _Float16* __restrict__ xo,
                              const float* __restrict__ cosb,
                              const float* __restrict__ sinb, int nheads, int total)
{
  int idx = blockIdx.x * blockDim.x + threadIdx.x;
  if (idx >= total) return;
  int d  = idx & 63;
  int h  = (idx >> 6) % nheads;
  int bs = idx / (64 * nheads);
  size_t base = ((size_t)bs * nheads + h) * HEADD;
  float c0 = cosb[(size_t)bs * HEADD + d];
  float s0 = sinb[(size_t)bs * HEADD + d];
  float c1 = cosb[(size_t)bs * HEADD + d + 64];
  float s1 = sinb[(size_t)bs * HEADD + d + 64];
  float x0 = x[base + d];
  float x1 = x[base + d + 64];
  xo[base + d]      = (_Float16)(x0 * c0 - x1 * s0);
  xo[base + d + 64] = (_Float16)(x1 * c1 + x0 * s1);
}

// ---------------------------------------------------------------------------
// Scores: logits = (q . k^T) * HD^-0.5, asym LSQ, causal mask -> attn buffer.
// ---------------------------------------------------------------------------
__global__ __launch_bounds__(256) void attn_scores_wmma(
    const _Float16* __restrict__ q, const _Float16* __restrict__ k,
    float* __restrict__ attn, const float* __restrict__ s_attn_ptr)
{
  __shared__ _Float16 Qs[16][136];
  __shared__ _Float16 Ks[8][16][136];
  int tid = threadIdx.x, lane = tid & 31, w = tid >> 5;
  int bhq = blockIdx.x;
  int qt  = bhq & 63;
  int bh  = bhq >> 6;
  int h   = bh % NHQ, b = bh / NHQ;
  int kvh = h / (NHQ / NKVH);
  int ktile = blockIdx.y * 8 + w;

  {                                      // Q tile 16x128: 1 b128 per thread
    int r = tid >> 4, c = (tid & 15) * 8;
    *reinterpret_cast<float4*>(&Qs[r][c]) = *reinterpret_cast<const float4*>(
        &q[((size_t)(b * S_LEN + qt * 16 + r) * NHQ + h) * HEADD + c]);
  }
  {                                      // wave-private K tile 16x128: 8 b128
    int krow0 = ktile * 16;
#pragma unroll
    for (int i = 0; i < 8; ++i) {
      int ch = i * 32 + lane;
      int r = ch >> 4, c = (ch & 15) * 8;
      *reinterpret_cast<float4*>(&Ks[w][r][c]) = *reinterpret_cast<const float4*>(
          &k[((size_t)(b * S_LEN + krow0 + r) * NKVH + kvh) * HEADD + c]);
    }
  }
  __syncthreads();

  v8f acc = {};
#pragma unroll
  for (int dstep = 0; dstep < 4; ++dstep) {
    v16h a  = frag_a_16x32(&Qs[0][dstep * 32], 136, lane);
    v16h bf = frag_b_32x16(&Ks[w][0][dstep * 32], 136, lane);
    acc = __builtin_amdgcn_wmma_f32_16x16x32_f16(false, a, false, bf,
                                                 (short)0, acc, false, false);
  }

  float s = *s_attn_ptr;
  const float scaling = 0.08838834764831845f;   // 128^-0.5
  int n     = ktile * 16 + (lane & 15);
  int mbase = qt * 16 + ((lane < 16) ? 0 : 8);
  float* rowbase = attn + (size_t)bh * S_LEN * S_LEN;
#pragma unroll
  for (int r = 0; r < 8; ++r) {
    int m = mbase + r;
    float v = lsq_fwd(acc[r] * scaling, s, 0);
    if (n > m) v = -3.4028234663852886e38f;     // causal mask (== + finfo.min)
    rowbase[(size_t)m * S_LEN + n] = v;
  }
}

// ---------------------------------------------------------------------------
// In-place rowwise softmax over S=1024 columns; one block per row.
// ---------------------------------------------------------------------------
__global__ __launch_bounds__(256) void softmax_rows(float* __restrict__ attn)
{
  __shared__ float red[256];
  float* row = attn + (size_t)blockIdx.x * S_LEN;
  int tid = threadIdx.x;
  float m = -3.4028234663852886e38f;
  for (int c = tid; c < S_LEN; c += 256) m = fmaxf(m, row[c]);
  red[tid] = m; __syncthreads();
  for (int st = 128; st > 0; st >>= 1) {
    if (tid < st) red[tid] = fmaxf(red[tid], red[tid + st]);
    __syncthreads();
  }
  m = red[0]; __syncthreads();
  float sum = 0.0f;
  for (int c = tid; c < S_LEN; c += 256) sum += __expf(row[c] - m);
  red[tid] = sum; __syncthreads();
  for (int st = 128; st > 0; st >>= 1) {
    if (tid < st) red[tid] += red[tid + st];
    __syncthreads();
  }
  float inv = 1.0f / red[0];
  for (int c = tid; c < S_LEN; c += 256) row[c] = __expf(row[c] - m) * inv;
}

// ---------------------------------------------------------------------------
// ctx_h = LSQ(attn @ v) (K = 1024, 32 WMMA steps). v fp16; probs fp32 packed
// to fp16 pairs (b32 LDS stores). V staged transposed in LDS.
// ---------------------------------------------------------------------------
__global__ __launch_bounds__(256) void attn_v_wmma(
    const float* __restrict__ attn, const _Float16* __restrict__ v,
    _Float16* __restrict__ ctxh, const float* __restrict__ s_after_ptr)
{
  __shared__ _Float16 Ps[16][40];
  __shared__ _Float16 Vt[128][40];       // [d][kvrow]
  int tid = threadIdx.x, lane = tid & 31, w = tid >> 5;
  int bhq = blockIdx.x;
  int qt  = bhq & 63;
  int bh  = bhq >> 6;
  int h   = bh % NHQ, b = bh / NHQ;
  int kvh = h / (NHQ / NKVH);
  const float* arow = attn + (size_t)bh * S_LEN * S_LEN;

  v8f acc = {};
  for (int k0 = 0; k0 < S_LEN; k0 += 32) {
    __syncthreads();
    {                                    // probs 16x32: 2 fp32 -> 1 b32 store
      int r = tid >> 4, c = (tid & 15) * 2;
      float p0 = arow[(size_t)(qt * 16 + r) * S_LEN + k0 + c];
      float p1 = arow[(size_t)(qt * 16 + r) * S_LEN + k0 + c + 1];
      union { _Float16 h[2]; unsigned int u; } pk;
      pk.h[0] = (_Float16)p0; pk.h[1] = (_Float16)p1;
      *reinterpret_cast<unsigned int*>(&Ps[r][c]) = pk.u;
    }
#pragma unroll
    for (int i = 0; i < 2; ++i) {        // V 32x128 fp16: b128 load, store ^T
      int ch = i * 256 + tid;
      int r = ch >> 4, c = (ch & 15) * 8;
      float4 raw = *reinterpret_cast<const float4*>(
          &v[((size_t)(b * S_LEN + k0 + r) * NKVH + kvh) * HEADD + c]);
      const _Float16* hp = reinterpret_cast<const _Float16*>(&raw);
#pragma unroll
      for (int j = 0; j < 8; ++j) Vt[c + j][r] = hp[j];
    }
    __syncthreads();
    v16h a  = frag_a_16x32(&Ps[0][0], 40, lane);
    v16h bf = frag_b_32x16(&Vt[w * 16][0], 40, lane);
    acc = __builtin_amdgcn_wmma_f32_16x16x32_f16(false, a, false, bf,
                                                 (short)0, acc, false, false);
  }

  float s = *s_after_ptr;
  int d     = w * 16 + (lane & 15);
  int mbase = qt * 16 + ((lane < 16) ? 0 : 8);
#pragma unroll
  for (int r = 0; r < 8; ++r) {
    int m = mbase + r;
    float val = lsq_fwd(acc[r], s, 0);
    ctxh[(size_t)(b * S_LEN + m) * (NHQ * HEADD) + h * HEADD + d] = (_Float16)val;
  }
}

// ---------------------------------------------------------------------------
extern "C" void kernel_launch(void* const* d_in, const int* in_sizes, int n_in,
                              void* d_out, int out_size, void* d_ws, size_t ws_size,
                              hipStream_t stream) {
  (void)in_sizes; (void)n_in; (void)out_size; (void)ws_size;
  const float* hidden = (const float*)d_in[0];
  const float* cosb   = (const float*)d_in[1];
  const float* sinb   = (const float*)d_in[2];
  /* d_in[3] attention_mask: causal, recomputed in-kernel */
  const float* Wq     = (const float*)d_in[4];
  const float* Wk     = (const float*)d_in[5];
  const float* Wv     = (const float*)d_in[6];
  const float* Wo     = (const float*)d_in[7];
  const float* s_q    = (const float*)d_in[8];
  const float* s_k    = (const float*)d_in[9];
  const float* s_v    = (const float*)d_in[10];
  const float* s_attn = (const float*)d_in[11];
  const float* s_aft  = (const float*)d_in[12];
  const float* s_o    = (const float*)d_in[13];

  const size_t NQ  = (size_t)BATCH * S_LEN * NHQ  * HEADD;  // 16.78M
  const size_t NKV = (size_t)BATCH * S_LEN * NKVH * HEADD;  //  4.19M
  const size_t NH_ = (size_t)BATCH * S_LEN * HDIM;          // 16.78M
  const size_t NW  = (size_t)HDIM * HDIM;                   // 16.78M
  const size_t NWk = (size_t)NKVH * HEADD * HDIM;           //  4.19M

  float* ws  = (float*)d_ws;
  float* qb  = ws;                 // [B*S, NH, HD] fp32 (pre-RoPE)
  float* kb  = qb + NQ;            // [B*S, KV, HD] fp32 (pre-RoPE)
  _Float16* hh  = (_Float16*)(kb + NKV);   // hidden fp16
  _Float16* wqh = hh  + NH_;
  _Float16* wkh = wqh + NW;
  _Float16* wvh = wkh + NWk;
  _Float16* woh = wvh + NWk;
  _Float16* qh  = woh + NW;        // q fp16 (post-RoPE)
  _Float16* kh  = qh  + NQ;        // k fp16 (post-RoPE)
  _Float16* vh  = kh  + NKV;       // v fp16 (quantized)
  _Float16* ctxh= vh  + NKV;       // attn@v fp16

  float* out  = (float*)d_out;                               // [B*S, 4096]
  float* attn = out + NQ;                                    // [B,NH,S,S]

  const int M = BATCH * S_LEN;   // 4096
  dim3 blk(256);

  // One-time fp32 -> fp16 staging of GEMM operands
  f32_to_f16_kernel<<<(int)(NH_ / 4 + 255) / 256, blk, 0, stream>>>(hidden, hh, (int)(NH_ / 4));
  f32_to_f16_kernel<<<(int)(NW  / 4 + 255) / 256, blk, 0, stream>>>(Wq, wqh, (int)(NW  / 4));
  f32_to_f16_kernel<<<(int)(NWk / 4 + 255) / 256, blk, 0, stream>>>(Wk, wkh, (int)(NWk / 4));
  f32_to_f16_kernel<<<(int)(NWk / 4 + 255) / 256, blk, 0, stream>>>(Wv, wvh, (int)(NWk / 4));
  f32_to_f16_kernel<<<(int)(NW  / 4 + 255) / 256, blk, 0, stream>>>(Wo, woh, (int)(NW  / 4));

  // Projections (sym LSQ fused). V goes straight to fp16 (only PV consumes it).
  gemm_h_wmma<<<dim3(M / 128, (NHQ  * HEADD) / 64), blk, 0, stream>>>(
      hh, wqh, qb, nullptr, M, NHQ * HEADD, HDIM, s_q, 1);
  gemm_h_wmma<<<dim3(M / 128, (NKVH * HEADD) / 64), blk, 0, stream>>>(
      hh, wkh, kb, nullptr, M, NKVH * HEADD, HDIM, s_k, 1);
  gemm_h_wmma<<<dim3(M / 128, (NKVH * HEADD) / 64), blk, 0, stream>>>(
      hh, wvh, nullptr, vh, M, NKVH * HEADD, HDIM, s_v, 1);

  // RoPE (after quant, matching reference order), emits fp16 operands
  int totq = M * NHQ * 64;
  rope_h_kernel<<<(totq + 255) / 256, blk, 0, stream>>>(qb, qh, cosb, sinb, NHQ, totq);
  int totk = M * NKVH * 64;
  rope_h_kernel<<<(totk + 255) / 256, blk, 0, stream>>>(kb, kh, cosb, sinb, NKVH, totk);

  // Scores -> quant -> mask (into mandatory attn output region)
  attn_scores_wmma<<<dim3(BATCH * NHQ * 64, 8), blk, 0, stream>>>(qh, kh, attn, s_attn);
  // Softmax in place
  softmax_rows<<<dim3(BATCH * NHQ * S_LEN), blk, 0, stream>>>(attn);
  // attn @ v (asym LSQ fused), fp16 ctx out
  attn_v_wmma<<<dim3(BATCH * NHQ * 64), blk, 0, stream>>>(attn, vh, ctxh, s_aft);
  // Output projection (sym LSQ fused)
  gemm_h_wmma<<<dim3(M / 128, HDIM / 64), blk, 0, stream>>>(
      ctxh, woh, out, nullptr, M, HDIM, HDIM, s_o, 1);
}
